// HONU_68968584839417
// MI455X (gfx1250) — matrix operand order repaired and checked by reading, not compiled, not run on hardware
//
#include <hip/hip_runtime.h>

typedef _Float16 h8   __attribute__((ext_vector_type(8)));
typedef _Float16 v16h __attribute__((ext_vector_type(16)));
typedef float    v8f  __attribute__((ext_vector_type(8)));

#define NPAIR  8128   // pairs (j<=k) over 127 features; = 254 * 32 (exact K tiling)
#define NFEATP 128    // 127 real features (bias + 126) padded with one zero column
#define BATCHN 256

// T(j) = index of first pair whose smaller element is j (combinations-with-replacement order)
__device__ __forceinline__ int triT(int j) { return (j * (255 - j)) >> 1; }

// --- Kernel 1: build x_full (f32, [256x128]) and pair products P (f16, [256x8128]) ---
__global__ void __launch_bounds__(256) honu_build_xfull_P(
    const float* __restrict__ x, float* __restrict__ xfull, _Float16* __restrict__ P)
{
    __shared__ float sx[NFEATP];
    const int b = blockIdx.x;
    const int tid = threadIdx.x;
    if (tid < 126)       sx[tid + 1] = x[b * 126 + tid];
    else if (tid == 126) sx[0]   = 1.0f;   // bias feature
    else if (tid == 127) sx[127] = 0.0f;   // zero pad column
    __syncthreads();
    if (tid < NFEATP) xfull[b * NFEATP + tid] = sx[tid];
    for (int p = tid; p < NPAIR; p += 256) {
        // invert p -> (j,k): analytic guess + fixup
        float disc = 65025.0f - 8.0f * (float)p;      // 255^2 - 8p
        int j = (int)((255.0f - sqrtf(disc)) * 0.5f);
        if (j < 0) j = 0; if (j > 126) j = 126;
        while (j < 126 && triT(j + 1) <= p) ++j;
        while (j > 0 && triT(j) > p) --j;
        int k = j + (p - triT(j));
        P[b * NPAIR + p] = (_Float16)(sx[j] * sx[k]);
    }
}

// --- Kernel 2: zero Mt (f16, [128 x 8128]) in 16-byte chunks ---
__global__ void __launch_bounds__(256) honu_zero(uint4* __restrict__ dst, int n16)
{
    int i = blockIdx.x * blockDim.x + threadIdx.x;
    int stride = gridDim.x * blockDim.x;
    uint4 z = {0u, 0u, 0u, 0u};
    for (; i < n16; i += stride) dst[i] = z;
}

// --- Kernel 3: scatter weights into Mt[i0, pair(i1,i2)] (unique slots, no atomics) ---
__global__ void __launch_bounds__(256) honu_scatter(
    const int* __restrict__ comb, const float* __restrict__ w,
    _Float16* __restrict__ Mt, int ncomb)
{
    int c = blockIdx.x * blockDim.x + threadIdx.x;
    if (c >= ncomb) return;
    int i0 = comb[3 * c + 0];
    int i1 = comb[3 * c + 1];
    int i2 = comb[3 * c + 2];
    int p = triT(i1) + (i2 - i1);
    Mt[i0 * NPAIR + p] = (_Float16)w[c];
}

__device__ __forceinline__ v16h honu_frag(h8 lo, h8 hi)
{
    v16h f;
#pragma unroll
    for (int t = 0; t < 8; ++t) { f[t] = lo[t]; f[t + 8] = hi[t]; }
    return f;
}

// --- Kernel 4: WMMA GEMM  Y[b,i] = sum_p P[b,p] * Mt[i,p] ---
// Register-blocked: each wave32 computes a 16x32 output strip (2 accumulators),
// sharing one A fragment across two B fragments per K-step.
// 16 M-tiles x 4 N-strips = 64 waves = 8 blocks x 8 waves.
__global__ void __launch_bounds__(256) honu_wmma_gemm(
    const _Float16* __restrict__ P, const _Float16* __restrict__ Mt,
    float* __restrict__ Y)
{
    const int lane = threadIdx.x & 31;
    const int wave = threadIdx.x >> 5;
    const int strip = blockIdx.x * 8 + wave;  // 0..63
    const int tm  = strip >> 2;               // 0..15 (batch tile)
    const int tnp = strip & 3;                // 0..3  (pair of feature tiles)
    const int r  = lane & 15;
    const int h  = lane >> 4;

    const _Float16* Arow  = P  + (size_t)(tm * 16 + r) * NPAIR;        // A row (batch)
    const _Float16* Brow0 = Mt + (size_t)(tnp * 32 + r) * NPAIR;       // B rows (features)
    const _Float16* Brow1 = Mt + (size_t)(tnp * 32 + 16 + r) * NPAIR;

    v8f acc0 = {}, acc1 = {};
#pragma unroll 2
    for (int kb = 0; kb < NPAIR; kb += 32) {
        // A fragment (16x32 f16): lane M=r; K = 16h+0..15 -> two b128 loads
        h8 alo = *(const h8*)(Arow + kb + 16 * h);
        h8 ahi = *(const h8*)(Arow + kb + 16 * h + 8);
        // B fragments (32x16 f16): lane N=r; K = 16h+0..15 -> two b128 loads each
        h8 b0lo = *(const h8*)(Brow0 + kb + 16 * h);
        h8 b0hi = *(const h8*)(Brow0 + kb + 16 * h + 8);
        h8 b1lo = *(const h8*)(Brow1 + kb + 16 * h);
        h8 b1hi = *(const h8*)(Brow1 + kb + 16 * h + 8);
        __builtin_prefetch(Arow  + kb + 256, 0, 1);  // gfx1250 global_prefetch_b8
        __builtin_prefetch(Brow0 + kb + 256, 0, 1);
        __builtin_prefetch(Brow1 + kb + 256, 0, 1);
        v16h a  = honu_frag(alo, ahi);
        v16h b0 = honu_frag(b0lo, b0hi);
        v16h b1 = honu_frag(b1lo, b1hi);
        acc0 = __builtin_amdgcn_wmma_f32_16x16x32_f16(
            false, a, false, b0, (short)0, acc0, false, false);
        acc1 = __builtin_amdgcn_wmma_f32_16x16x32_f16(
            false, a, false, b1, (short)0, acc1, false, false);
    }
    // C/D layout: VGPR rr -> M = rr + 8h, N = r
    float* Yout = Y + (size_t)(tm * 16 + 8 * h) * NFEATP + tnp * 32 + r;
#pragma unroll
    for (int rr = 0; rr < 8; ++rr) {
        Yout[rr * NFEATP]      = acc0[rr];
        Yout[rr * NFEATP + 16] = acc1[rr];
    }
}

// --- Kernel 5: out[b] = sum_i xfull[b,i] * Y[b,i]; one wave32 per batch row ---
__global__ void __launch_bounds__(256) honu_final(
    const float* __restrict__ xfull, const float* __restrict__ Y,
    float* __restrict__ out)
{
    const int lane = threadIdx.x & 31;
    const int wave = threadIdx.x >> 5;
    const int b = blockIdx.x * 8 + wave;
    float s = 0.0f;
#pragma unroll
    for (int t = 0; t < 4; ++t) {
        int i = lane + 32 * t;
        s += xfull[b * NFEATP + i] * Y[b * NFEATP + i];
    }
#pragma unroll
    for (int off = 16; off > 0; off >>= 1) s += __shfl_xor(s, off, 32);
    if (lane == 0) out[b] = s;
}

extern "C" void kernel_launch(void* const* d_in, const int* in_sizes, int n_in,
                              void* d_out, int out_size, void* d_ws, size_t ws_size,
                              hipStream_t stream)
{
    const float* x    = (const float*)d_in[0];   // [256, 126] f32
    const float* w    = (const float*)d_in[1];   // [349504]   f32
    const int*   comb = (const int*)d_in[2];     // [349504,3] int32
    float* out = (float*)d_out;                  // [256] f32
    const int ncomb = in_sizes[1];               // 349504

    // workspace layout (256B-aligned offsets)
    char* ws = (char*)d_ws;
    float*    xfull = (float*)(ws + 0);          // 256*128*4   = 131072
    _Float16* P     = (_Float16*)(ws + 131072);  // 256*8128*2  = 4161536
    _Float16* Mt    = (_Float16*)(ws + 4292608); // 128*8128*2  = 2080768
    float*    Y     = (float*)(ws + 6373376);    // 256*128*4   = 131072

    honu_build_xfull_P<<<BATCHN, 256, 0, stream>>>(x, xfull, P);

    const int n16 = (NFEATP * NPAIR) / 8;        // b128 chunks of Mt
    honu_zero<<<256, 256, 0, stream>>>((uint4*)Mt, n16);

    honu_scatter<<<(ncomb + 255) / 256, 256, 0, stream>>>(comb, w, Mt, ncomb);

    honu_wmma_gemm<<<8, 256, 0, stream>>>(P, Mt, Y);

    honu_final<<<32, 256, 0, stream>>>(xfull, Y, out);
}
